// SelfAttention_43026982371766
// MI455X (gfx1250) — matrix operand order repaired and checked by reading, not compiled, not run on hardware
//
#include <hip/hip_runtime.h>
#include <cstdint>

// ---------------------------------------------------------------------------
// Multi-head self-attention, MI455X (gfx1250), wave32 + WMMA f16->f32,
// async-to-LDS (ASYNCcnt) staging for weight / K / V tiles.
//
//   1) x (f32) -> x_h (f16);  W{q,k,v,o} (f32) -> pair-packed f16 B-layout
//   2) fused QKV GEMM: 128x64 block tile, 32x32 per wave (4 WMMA/k-step),
//      B tile async-staged to LDS, double buffered, branch-free inner loop
//   3) flash attention per (b,h): K/V chunks async-staged to LDS; QK^T WMMA,
//      online softmax (half-wave shfl_xor reductions), P via LDS bounce,
//      P@V WMMA
//   4) output projection GEMM + bias -> f32 d_out
// ---------------------------------------------------------------------------

typedef __attribute__((ext_vector_type(16))) _Float16 v16h;
typedef __attribute__((ext_vector_type(8)))  float    v8f;

#define EMBED    1024
#define HEADS    16
#define HEAD_DIM 64
#define BATCH    2
#define SEQ      2048
#define TOKENS   (BATCH * SEQ)         // 4096

// ---- workspace layout (bytes) ---------------------------------------------
#define WS_XH   ((size_t)0)                              // 8 MB f16
#define WS_WQP  (WS_XH  + (size_t)TOKENS * EMBED * 2)
#define WS_WKP  (WS_WQP + (size_t)EMBED/2 * EMBED * 4)   // 2 MB each (u32)
#define WS_WVP  (WS_WKP + (size_t)EMBED/2 * EMBED * 4)
#define WS_WOP  (WS_WVP + (size_t)EMBED/2 * EMBED * 4)
#define WS_QH   (WS_WOP + (size_t)EMBED/2 * EMBED * 4)   // per-head f16, 8 MB
#define WS_KH   (WS_QH  + (size_t)TOKENS * EMBED * 2)
#define WS_VP   (WS_KH  + (size_t)TOKENS * EMBED * 2)    // pair-packed f16
#define WS_AH   (WS_VP  + (size_t)TOKENS * EMBED * 2)    // attn out f16

// padded LDS row strides (u32 units) -> conflict-free frag reads, 16B-aligned
#define BSTRIDE 68   // B/V tiles: 16 rows x 64 u32 payload
#define KSTRIDE 36   // K tile:   32 rows x 32 u32 payload

static __device__ inline v8f wmma_f16(v16h a, v16h b, v8f c) {
    return __builtin_amdgcn_wmma_f32_16x16x32_f16(false, a, false, b,
                                                  (short)0, c, false, false);
}

// low 32 bits of a generic pointer into LDS == wave-relative LDS byte address
static __device__ inline uint32_t lds_off(const void* p) {
    return (uint32_t)(uintptr_t)p;
}

// CDNA5 async DMA: each lane copies 16B global -> LDS, tracked by ASYNCcnt
static __device__ inline void async_b128(uint32_t lds_addr,
                                         const uint32_t* __restrict__ g) {
    asm volatile("global_load_async_to_lds_b128 %0, %1, off"
                 :: "v"(lds_addr), "v"(g) : "memory");
}

// ---- fragment loaders (contiguous k order k = 16*half + s, wave32) -------------------------
// A 16x32 f16 (row-major, stride in u32): lane = half*16+m,
//   VGPR v: k = k0 + half*16 + 2*v
static __device__ inline v16h load_a_frag(const uint32_t* __restrict__ base,
                                          int row0, int stride_u32, int k0) {
    const int lane = threadIdx.x & 31;
    const int half = lane >> 4;
    const int m    = lane & 15;
    union { uint32_t u[8]; v16h x; } f;
#pragma unroll
    for (int v = 0; v < 8; ++v) {
        int k = k0 + half * 16 + 2 * v;
        f.u[v] = base[(size_t)(row0 + m) * stride_u32 + (k >> 1)];
    }
    return f.x;
}

// B 32x16 f16 from LDS-staged pair-packed tile (row = k-pair, BSTRIDE u32)
static __device__ inline v16h load_b_frag_lds(const uint32_t* __restrict__ lb,
                                              int n0) {
    const int lane = threadIdx.x & 31;
    const int half = lane >> 4;
    const int n    = n0 + (lane & 15);
    union { uint32_t u[8]; v16h x; } f;
#pragma unroll
    for (int v = 0; v < 8; ++v)
        f.u[v] = lb[(half * 8 + v) * BSTRIDE + n];
    return f.x;
}

// B fragment of K^T from LDS-staged K chunk (32 keys x 32 u32, KSTRIDE pad)
static __device__ inline v16h load_kT_lds(const uint32_t* __restrict__ lk,
                                          int d0, int key0) {
    const int lane = threadIdx.x & 31;
    const int half = lane >> 4;
    const int key  = key0 + (lane & 15);
    union { uint32_t u[8]; v16h x; } f;
#pragma unroll
    for (int v = 0; v < 8; ++v)
        f.u[v] = lk[key * KSTRIDE + (d0 >> 1) + half * 8 + v];
    return f.x;
}

// B fragment of V from LDS-staged pair-packed chunk (16 rows x 64 u32)
static __device__ inline v16h load_v_lds(const uint32_t* __restrict__ lv,
                                         int d0) {
    const int lane = threadIdx.x & 31;
    const int half = lane >> 4;
    const int d    = d0 + (lane & 15);
    union { uint32_t u[8]; v16h x; } f;
#pragma unroll
    for (int v = 0; v < 8; ++v)
        f.u[v] = lv[(half * 8 + v) * BSTRIDE + d];
    return f.x;
}

// ---------------------------------------------------------------------------
// 1) conversions
// ---------------------------------------------------------------------------
__global__ void f32_to_f16_kernel(const float* __restrict__ src,
                                  _Float16* __restrict__ dst, int n) {
    int i = blockIdx.x * blockDim.x + threadIdx.x;
    if (i < n) dst[i] = (_Float16)src[i];
}

// Wp[(k>>1)*1024 + n] = pack_f16(W[k][n], W[k+1][n])
__global__ void pack_weight_kernel(const float* __restrict__ W,
                                   uint32_t* __restrict__ Wp) {
    int i = blockIdx.x * blockDim.x + threadIdx.x;
    int kk = i >> 10, n = i & 1023;
    union { uint32_t u; _Float16 h[2]; } cv;
    cv.h[0] = (_Float16)W[(size_t)(2 * kk) * EMBED + n];
    cv.h[1] = (_Float16)W[(size_t)(2 * kk + 1) * EMBED + n];
    Wp[i] = cv.u;
}

// ---------------------------------------------------------------------------
// 2) fused QKV GEMM. Block tile 128x64, 8 waves, 32x32 per wave.
//    B tile (16 k-pairs x 64 u32) async-staged, double buffered.
// ---------------------------------------------------------------------------
__global__ __launch_bounds__(256)
void qkv_gemm_kernel(const _Float16* __restrict__ x_h,
                     const uint32_t* __restrict__ wq_p,
                     const uint32_t* __restrict__ wk_p,
                     const uint32_t* __restrict__ wv_p,
                     const float* __restrict__ bq,
                     const float* __restrict__ bk,
                     const float* __restrict__ bv,
                     _Float16* __restrict__ q_h,
                     _Float16* __restrict__ k_h,
                     uint16_t* __restrict__ v_p) {
    __shared__ __align__(16) uint32_t ldsB[2][16 * BSTRIDE];

    const int which = blockIdx.z;
    const uint32_t* wp  = (which == 0) ? wq_p : (which == 1) ? wk_p : wv_p;
    const float*    bia = (which == 0) ? bq   : (which == 1) ? bk   : bv;

    const int t = threadIdx.x, w = t >> 5, lane = t & 31;
    const int mw = w & 3, nw = w >> 2;
    const int row0 = blockIdx.x * 128 + mw * 32;
    const int nb0  = blockIdx.y * 64;

    // stage mapping: thread -> (k-pair row, 4-u32 column group)
    const int sr = t >> 4, sc4 = (t & 15) << 2;
    const uint32_t* gB = wp + (size_t)sr * EMBED + nb0 + sc4;
    const uint32_t lB[2] = { lds_off(&ldsB[0][sr * BSTRIDE + sc4]),
                             lds_off(&ldsB[1][sr * BSTRIDE + sc4]) };

    const uint32_t* a32 = (const uint32_t*)x_h;
    v8f acc[2][2] = {{v8f{}, v8f{}}, {v8f{}, v8f{}}};

    auto step = [&](int cur, int kk) {
        __syncthreads();
        v16h a0 = load_a_frag(a32, row0,      EMBED / 2, kk);
        v16h a1 = load_a_frag(a32, row0 + 16, EMBED / 2, kk);
        v16h b0 = load_b_frag_lds(&ldsB[cur][0], nw * 32);
        v16h b1 = load_b_frag_lds(&ldsB[cur][0], nw * 32 + 16);
        acc[0][0] = wmma_f16(a0, b0, acc[0][0]);
        acc[0][1] = wmma_f16(a0, b1, acc[0][1]);
        acc[1][0] = wmma_f16(a1, b0, acc[1][0]);
        acc[1][1] = wmma_f16(a1, b1, acc[1][1]);
        __syncthreads();
    };

    async_b128(lB[0], gB);                       // prologue: k0 = 0
    int k0 = 0;
#pragma unroll 2
    for (; k0 < EMBED - 32; k0 += 32) {          // branch-free steady state
        const int cur = (k0 >> 5) & 1;
        async_b128(lB[cur ^ 1], gB + (size_t)((k0 + 32) >> 1) * EMBED);
        asm volatile("s_wait_asynccnt 0x1" ::: "memory");
        step(cur, k0);
    }
    asm volatile("s_wait_asynccnt 0x0" ::: "memory");
    step((k0 >> 5) & 1, k0);                     // peeled last iteration

    // epilogue: C layout M=(lane>>4)*8+j, N=lane&15
    const int Nl = lane & 15, Mb = (lane >> 4) * 8;
#pragma unroll
    for (int mi = 0; mi < 2; ++mi) {
#pragma unroll
        for (int j = 0; j < 8; ++j) {
            int token = row0 + mi * 16 + Mb + j;
            int s  = token & (SEQ - 1);
            int bb = token >> 11;
#pragma unroll
            for (int ni = 0; ni < 2; ++ni) {
                int   n   = nb0 + nw * 32 + ni * 16 + Nl;
                float val = acc[mi][ni][j] + bia[n];
                int h = n >> 6, d = n & 63;
                size_t hb = ((size_t)(bb * HEADS + h)) << 17;   // * S*HEAD_DIM
                if (which == 0) {
                    q_h[hb + (size_t)s * 64 + d] = (_Float16)(val * 0.125f);
                } else if (which == 1) {
                    k_h[hb + (size_t)s * 64 + d] = (_Float16)val;
                } else {
                    union { uint16_t u; _Float16 h; } cv; cv.h = (_Float16)val;
                    v_p[hb + ((size_t)(s >> 1) * 64 + d) * 2 + (s & 1)] = cv.u;
                }
            }
        }
    }
}

// ---------------------------------------------------------------------------
// 3) flash attention: grid (B*H, S/128); 8 waves x 16 query rows.
//    K/V chunks (32 keys) async-staged to LDS, double buffered.
// ---------------------------------------------------------------------------
__global__ __launch_bounds__(256)
void flash_attn_kernel(const _Float16* __restrict__ q_h,
                       const _Float16* __restrict__ k_h,
                       const uint32_t* __restrict__ v_p32,
                       _Float16* __restrict__ attn_h) {
    __shared__ __align__(16) uint32_t ldsK[2][32 * KSTRIDE];
    __shared__ __align__(16) uint32_t ldsV[2][16 * BSTRIDE];
    __shared__ __align__(16) _Float16 lds_p[8][16][34];   // stride 17 u32

    const int bh   = blockIdx.x;
    const int t    = threadIdx.x;
    const int w    = t >> 5;
    const int lane = t & 31;
    const int q0   = blockIdx.y * 128 + w * 16;

    const size_t head_halves = (size_t)bh << 17;
    const uint32_t* Q32 = (const uint32_t*)(q_h + head_halves);
    const uint32_t* K32 = (const uint32_t*)(k_h + head_halves);
    const uint32_t* V32 = v_p32 + (head_halves >> 1);

    // stage maps: K chunk = 32 keys x 32 u32; V chunk = 16 pair-rows x 64 u32
    const int kr = t >> 3, kc4 = (t & 7) << 2;
    const int vr = t >> 4, vc4 = (t & 15) << 2;
    const uint32_t* gK = K32 + (size_t)kr * 32 + kc4;   // += kc*32 per chunk
    const uint32_t* gV = V32 + (size_t)vr * 64 + vc4;   // += kc*32 per chunk
    const uint32_t lK[2] = { lds_off(&ldsK[0][kr * KSTRIDE + kc4]),
                             lds_off(&ldsK[1][kr * KSTRIDE + kc4]) };
    const uint32_t lV[2] = { lds_off(&ldsV[0][vr * BSTRIDE + vc4]),
                             lds_off(&ldsV[1][vr * BSTRIDE + vc4]) };

    // Q strip 16x64 -> two resident A fragments (Q pre-scaled by 1/sqrt(d))
    v16h qa0 = load_a_frag(Q32, q0, 32, 0);
    v16h qa1 = load_a_frag(Q32, q0, 32, 32);

    v8f acc[4] = {v8f{}, v8f{}, v8f{}, v8f{}};
    float mrun[8], lrun[8];
#pragma unroll
    for (int j = 0; j < 8; ++j) { mrun[j] = -3.0e38f; lrun[j] = 0.f; }

    auto chunk = [&](int cur) {
        __syncthreads();
        // ---- scores S = (Q/sqrt(d)) K^T : 16x32 (two 16x16 tiles) ----
        v8f sc0 = {}, sc1 = {};
        {
            v16h kb0 = load_kT_lds(&ldsK[cur][0], 0,  0);
            v16h kb1 = load_kT_lds(&ldsK[cur][0], 0,  16);
            sc0 = wmma_f16(qa0, kb0, sc0);
            sc1 = wmma_f16(qa0, kb1, sc1);
            v16h kb2 = load_kT_lds(&ldsK[cur][0], 32, 0);
            v16h kb3 = load_kT_lds(&ldsK[cur][0], 32, 16);
            sc0 = wmma_f16(qa1, kb2, sc0);
            sc1 = wmma_f16(qa1, kb3, sc1);
        }

        // ---- online softmax (row j lives in this lane's 16-lane half) ----
#pragma unroll
        for (int j = 0; j < 8; ++j) {
            float mx = fmaxf(sc0[j], sc1[j]);
            mx = fmaxf(mx, __shfl_xor(mx, 1));
            mx = fmaxf(mx, __shfl_xor(mx, 2));
            mx = fmaxf(mx, __shfl_xor(mx, 4));
            mx = fmaxf(mx, __shfl_xor(mx, 8));
            float mnew  = fmaxf(mrun[j], mx);
            float scale = __expf(mrun[j] - mnew);
            float e0 = __expf(sc0[j] - mnew);
            float e1 = __expf(sc1[j] - mnew);
            float sm = e0 + e1;
            sm += __shfl_xor(sm, 1);
            sm += __shfl_xor(sm, 2);
            sm += __shfl_xor(sm, 4);
            sm += __shfl_xor(sm, 8);
            lrun[j] = lrun[j] * scale + sm;
            mrun[j] = mnew;
            acc[0][j] *= scale; acc[1][j] *= scale;
            acc[2][j] *= scale; acc[3][j] *= scale;
            int row = (lane >> 4) * 8 + j;
            lds_p[w][row][(lane & 15)]      = (_Float16)e0;
            lds_p[w][row][16 + (lane & 15)] = (_Float16)e1;
        }
        // wave-private LDS RAW: drain DS counter before re-reading P
        asm volatile("s_wait_dscnt 0x0" ::: "memory");

        v16h pa = load_a_frag((const uint32_t*)&lds_p[w][0][0], 0, 17, 0);
        // ---- O += P @ V (four 16x16 d-tiles) ----
#pragma unroll
        for (int dt = 0; dt < 4; ++dt) {
            v16h vb = load_v_lds(&ldsV[cur][0], dt * 16);
            acc[dt] = wmma_f16(pa, vb, acc[dt]);
        }
        __syncthreads();
    };

    async_b128(lK[0], gK);
    async_b128(lV[0], gV);
    int kc = 0;
#pragma unroll 2
    for (; kc < SEQ - 32; kc += 32) {            // branch-free steady state
        const int cur = (kc >> 5) & 1;
        async_b128(lK[cur ^ 1], gK + (size_t)(kc + 32) * 32);
        async_b128(lV[cur ^ 1], gV + (size_t)(kc + 32) * 32);
        asm volatile("s_wait_asynccnt 0x2" ::: "memory");
        chunk(cur);
    }
    asm volatile("s_wait_asynccnt 0x0" ::: "memory");
    chunk((kc >> 5) & 1);                        // peeled last chunk

    // ---- normalize + write (B,S,E) f16 for output projection ----
    const int b = bh >> 4, h = bh & 15;
#pragma unroll
    for (int j = 0; j < 8; ++j) {
        float inv = 1.0f / lrun[j];
        int row = (lane >> 4) * 8 + j;
        size_t tok = (size_t)b * SEQ + q0 + row;
#pragma unroll
        for (int dt = 0; dt < 4; ++dt) {
            int d = dt * 16 + (lane & 15);
            attn_h[tok * EMBED + h * 64 + d] = (_Float16)(acc[dt][j] * inv);
        }
    }
}

// ---------------------------------------------------------------------------
// 4) output projection: out = attn @ Wo + bo  (f32), same GEMM core
// ---------------------------------------------------------------------------
__global__ __launch_bounds__(256)
void out_proj_kernel(const _Float16* __restrict__ attn_h,
                     const uint32_t* __restrict__ wo_p,
                     const float* __restrict__ bo,
                     float* __restrict__ out) {
    __shared__ __align__(16) uint32_t ldsB[2][16 * BSTRIDE];

    const int t = threadIdx.x, w = t >> 5, lane = t & 31;
    const int mw = w & 3, nw = w >> 2;
    const int row0 = blockIdx.x * 128 + mw * 32;
    const int nb0  = blockIdx.y * 64;

    const int sr = t >> 4, sc4 = (t & 15) << 2;
    const uint32_t* gB = wo_p + (size_t)sr * EMBED + nb0 + sc4;
    const uint32_t lB[2] = { lds_off(&ldsB[0][sr * BSTRIDE + sc4]),
                             lds_off(&ldsB[1][sr * BSTRIDE + sc4]) };

    const uint32_t* a32 = (const uint32_t*)attn_h;
    v8f acc[2][2] = {{v8f{}, v8f{}}, {v8f{}, v8f{}}};

    auto step = [&](int cur, int kk) {
        __syncthreads();
        v16h a0 = load_a_frag(a32, row0,      EMBED / 2, kk);
        v16h a1 = load_a_frag(a32, row0 + 16, EMBED / 2, kk);
        v16h b0 = load_b_frag_lds(&ldsB[cur][0], nw * 32);
        v16h b1 = load_b_frag_lds(&ldsB[cur][0], nw * 32 + 16);
        acc[0][0] = wmma_f16(a0, b0, acc[0][0]);
        acc[0][1] = wmma_f16(a0, b1, acc[0][1]);
        acc[1][0] = wmma_f16(a1, b0, acc[1][0]);
        acc[1][1] = wmma_f16(a1, b1, acc[1][1]);
        __syncthreads();
    };

    async_b128(lB[0], gB);
    int k0 = 0;
#pragma unroll 2
    for (; k0 < EMBED - 32; k0 += 32) {
        const int cur = (k0 >> 5) & 1;
        async_b128(lB[cur ^ 1], gB + (size_t)((k0 + 32) >> 1) * EMBED);
        asm volatile("s_wait_asynccnt 0x1" ::: "memory");
        step(cur, k0);
    }
    asm volatile("s_wait_asynccnt 0x0" ::: "memory");
    step((k0 >> 5) & 1, k0);

    const int Nl = lane & 15, Mb = (lane >> 4) * 8;
#pragma unroll
    for (int mi = 0; mi < 2; ++mi) {
#pragma unroll
        for (int j = 0; j < 8; ++j) {
            size_t token = row0 + mi * 16 + Mb + j;
#pragma unroll
            for (int ni = 0; ni < 2; ++ni) {
                int n = nb0 + nw * 32 + ni * 16 + Nl;
                out[token * EMBED + n] = acc[mi][ni][j] + bo[n];
            }
        }
    }
}

// ---------------------------------------------------------------------------
extern "C" void kernel_launch(void* const* d_in, const int* in_sizes, int n_in,
                              void* d_out, int out_size, void* d_ws, size_t ws_size,
                              hipStream_t stream) {
    (void)in_sizes; (void)n_in; (void)out_size; (void)ws_size;
    const float* x  = (const float*)d_in[0];
    const float* Wq = (const float*)d_in[1];
    const float* bq = (const float*)d_in[2];
    const float* Wk = (const float*)d_in[3];
    const float* bk = (const float*)d_in[4];
    const float* Wv = (const float*)d_in[5];
    const float* bv = (const float*)d_in[6];
    const float* Wo = (const float*)d_in[7];
    const float* bo = (const float*)d_in[8];
    float* out = (float*)d_out;

    char* ws = (char*)d_ws;
    _Float16* x_h  = (_Float16*)(ws + WS_XH);
    uint32_t* wq_p = (uint32_t*)(ws + WS_WQP);
    uint32_t* wk_p = (uint32_t*)(ws + WS_WKP);
    uint32_t* wv_p = (uint32_t*)(ws + WS_WVP);
    uint32_t* wo_p = (uint32_t*)(ws + WS_WOP);
    _Float16* q_h  = (_Float16*)(ws + WS_QH);
    _Float16* k_h  = (_Float16*)(ws + WS_KH);
    uint16_t* v_p  = (uint16_t*)(ws + WS_VP);
    _Float16* a_h  = (_Float16*)(ws + WS_AH);

    f32_to_f16_kernel<<<(TOKENS * EMBED) / 256, 256, 0, stream>>>(x, x_h, TOKENS * EMBED);
    const int PACKN = (EMBED / 2) * EMBED;
    pack_weight_kernel<<<PACKN / 256, 256, 0, stream>>>(Wq, wq_p);
    pack_weight_kernel<<<PACKN / 256, 256, 0, stream>>>(Wk, wk_p);
    pack_weight_kernel<<<PACKN / 256, 256, 0, stream>>>(Wv, wv_p);
    pack_weight_kernel<<<PACKN / 256, 256, 0, stream>>>(Wo, wo_p);

    qkv_gemm_kernel<<<dim3(TOKENS / 128, EMBED / 64, 3), 256, 0, stream>>>(
        x_h, wq_p, wk_p, wv_p, bq, bk, bv, q_h, k_h, v_p);

    flash_attn_kernel<<<dim3(BATCH * HEADS, SEQ / 128), 256, 0, stream>>>(
        q_h, k_h, (const uint32_t*)v_p, a_h);

    out_proj_kernel<<<dim3(TOKENS / 128, EMBED / 64), 256, 0, stream>>>(
        a_h, wo_p, bo, out);
}
